// MultiTaskGridTransformer_54065048322319
// MI455X (gfx1250) — matrix operand order repaired and checked by reading, not compiled, not run on hardware
//
#include <hip/hip_runtime.h>
#include <hip/hip_bf16.h>
#include <math.h>

// ---------------------------------------------------------------------------
// MI455X / gfx1250 multi-task grid transformer forward.
// f16-native WMMA GEMMs + fused flash-attention (no S x S materialization).
// Residual stream and LN inputs stay f32; all matrix math via
// v_wmma_f32_16x16x32_f16 (wave32).
// ---------------------------------------------------------------------------

typedef __attribute__((ext_vector_type(16))) _Float16 v16h;
typedef __attribute__((ext_vector_type(8)))  _Float16 v8h;
typedef __attribute__((ext_vector_type(4)))  _Float16 v4h;
typedef __attribute__((ext_vector_type(8)))  float    v8f;

#define S_LEN 4096
#define DMODEL 512

// ---------------- wave helpers (wave32) ----------------
__device__ __forceinline__ float wave_sum(float v) {
#pragma unroll
    for (int off = 16; off > 0; off >>= 1) v += __shfl_xor(v, off, 32);
    return v;
}
__device__ __forceinline__ float wave_max(float v) {
#pragma unroll
    for (int off = 16; off > 0; off >>= 1) v = fmaxf(v, __shfl_xor(v, off, 32));
    return v;
}
// reductions within each 16-lane group (rows of a WMMA C tile)
__device__ __forceinline__ float half_sum(float v) {
#pragma unroll
    for (int off = 8; off > 0; off >>= 1) v += __shfl_xor(v, off, 32);
    return v;
}
__device__ __forceinline__ float half_max(float v) {
#pragma unroll
    for (int off = 8; off > 0; off >>= 1) v = fmaxf(v, __shfl_xor(v, off, 32));
    return v;
}

// f16 fragment: elems 0..7 <- p0[0..7], elems 8..15 <- p1[0..7]
__device__ __forceinline__ v16h load_fragh(const _Float16* p0, const _Float16* p1) {
    v8h x = *(const v8h*)p0;
    v8h y = *(const v8h*)p1;
    v16h r;
#pragma unroll
    for (int i = 0; i < 8; ++i) { r[i] = x[i]; r[8 + i] = y[i]; }
    return r;
}

__device__ __forceinline__ v8f wmma16(v16h a, v16h b, v8f c) {
    return __builtin_amdgcn_wmma_f32_16x16x32_f16(false, a, false, b,
                                                  (short)0, c, false, false);
}

// ---------------------------------------------------------------------------
// GEMM:  C[M,N] = act(alpha * (A @ B^T) + bias + resid)
//   A : [M,K] f16 row-major, lda ; B : [N,K] f16 row-major, ldb
//   resid f32 ; output to Cf (f32) and/or Ch (f16), same ldc.
//   act: 0=none 1=relu 2=gelu(erf)
// Wave tile: 32 rows x 64 cols (2 A frags x 4 B tiles -> 8 WMMA / K-step).
// Block 256 thr = 8 waves stacked in M -> 256 x 64 per block.
// Requires M%256==0, N%64==0, K%32==0.
// ---------------------------------------------------------------------------
__global__ void __launch_bounds__(256)
gemm_nt_wmma_h(const _Float16* __restrict__ A, int lda,
               const _Float16* __restrict__ B, int ldb,
               const float* __restrict__ bias,
               const float* __restrict__ resid, int ldr,
               float* __restrict__ Cf, _Float16* __restrict__ Ch, int ldc,
               int M, int N, int K, float alpha, int act)
{
    const int lane = threadIdx.x & 31;
    const int wave = threadIdx.x >> 5;
    const int wm   = (blockIdx.y * 8 + wave) * 32;
    const int wn   = blockIdx.x * 64;
    if (wm >= M || wn >= N) return;
    const int hs  = lane >> 4;
    const int l15 = lane & 15;

    v8f acc[2][4];
#pragma unroll
    for (int i = 0; i < 2; ++i)
#pragma unroll
        for (int j = 0; j < 4; ++j)
#pragma unroll
            for (int r = 0; r < 8; ++r) acc[i][j][r] = 0.0f;

    const _Float16* arow0 = A + (size_t)(wm + l15) * lda;
    const _Float16* arow1 = A + (size_t)(wm + 16 + l15) * lda;
    const _Float16* brow[4];
#pragma unroll
    for (int j = 0; j < 4; ++j)
        brow[j] = B + (size_t)(wn + j * 16 + l15) * ldb;

    for (int k0 = 0; k0 < K; k0 += 32) {
        v16h a0 = load_fragh(arow0 + k0 + hs * 16, arow0 + k0 + hs * 16 + 8);
        v16h a1 = load_fragh(arow1 + k0 + hs * 16, arow1 + k0 + hs * 16 + 8);
#pragma unroll
        for (int j = 0; j < 4; ++j) {
            v16h b = *(const v16h*)(brow[j] + k0 + hs * 16);
            acc[0][j] = wmma16(a0, b, acc[0][j]);
            acc[1][j] = wmma16(a1, b, acc[1][j]);
        }
    }

#pragma unroll
    for (int i = 0; i < 2; ++i) {
#pragma unroll
        for (int j = 0; j < 4; ++j) {
            const int n = wn + j * 16 + l15;
            const float bb = bias ? bias[n] : 0.0f;
#pragma unroll
            for (int r = 0; r < 8; ++r) {
                const int mrow = wm + i * 16 + hs * 8 + r;   // C/D: M = r + 8*half
                float v = acc[i][j][r] * alpha + bb;
                if (resid) v += resid[(size_t)mrow * ldr + n];
                if (act == 1)      v = fmaxf(v, 0.0f);
                else if (act == 2) v = 0.5f * v * (1.0f + erff(v * 0.70710678118654752f));
                if (Cf) Cf[(size_t)mrow * ldc + n] = v;
                if (Ch) Ch[(size_t)mrow * ldc + n] = (_Float16)v;
            }
        }
    }
}

// ---------------------------------------------------------------------------
// Fused flash attention. One wave = 16 query rows; key loop steps by 32 so
// P@V is a single 16x16x32 WMMA per output tile. Online softmax in registers;
// P converted C-layout -> A-layout through a 1KB wave-private LDS tile.
//   QKV : [S, 3D] f16 (Q at col 0, K at col D, V unused here)
//   VT  : [D, S]  f16 (V transposed; head h rows at h*HD)
//   O   : [S, D]  f16 (head h cols at h*HD)
// grid: (S/128, nheads), block 256 (8 waves).
// ---------------------------------------------------------------------------
template <int HD>
__global__ void __launch_bounds__(256)
flash_attn_k(const _Float16* __restrict__ QKV,
             const _Float16* __restrict__ VT,
             _Float16* __restrict__ O,
             float alpha)
{
    constexpr int NC = HD / 32;    // 32-dim chunks for Q.K^T
    constexpr int NT = HD / 16;    // 16-col output tiles for P@V
    const int lane = threadIdx.x & 31;
    const int wave = threadIdx.x >> 5;
    const int h    = blockIdx.y;
    const int q0   = (blockIdx.x * 8 + wave) * 16;
    const int hs   = lane >> 4;
    const int l15  = lane & 15;
    const int LD   = 3 * DMODEL;

    __shared__ _Float16 pbuf[8][16 * 32 + 16];
    _Float16* pw = pbuf[wave];

    const _Float16* qrow = QKV + (size_t)(q0 + l15) * LD + h * HD;
    const _Float16* Kb   = QKV + DMODEL + h * HD;
    const _Float16* Vb   = VT + (size_t)h * HD * S_LEN;

    v8f oacc[NT];
#pragma unroll
    for (int t = 0; t < NT; ++t)
#pragma unroll
        for (int r = 0; r < 8; ++r) oacc[t][r] = 0.0f;

    float rmax[8], rsum[8];
#pragma unroll
    for (int r = 0; r < 8; ++r) { rmax[r] = -3.0e38f; rsum[r] = 0.0f; }

    for (int k0 = 0; k0 < S_LEN; k0 += 32) {
        // ---- scores: two 16x16 tiles over keys [k0, k0+32) ----
        v8f sc0, sc1;
#pragma unroll
        for (int r = 0; r < 8; ++r) { sc0[r] = 0.0f; sc1[r] = 0.0f; }
        const _Float16* krow0 = Kb + (size_t)(k0 + l15) * LD;
        const _Float16* krow1 = krow0 + (size_t)16 * LD;
#pragma unroll
        for (int c = 0; c < NC; ++c) {
            const int d0 = c * 32;
            v16h qf  = load_fragh(qrow + d0 + hs * 16, qrow + d0 + hs * 16 + 8);
            v16h kf0 = *(const v16h*)(krow0 + d0 + hs * 16);
            v16h kf1 = *(const v16h*)(krow1 + d0 + hs * 16);
            sc0 = wmma16(qf, kf0, sc0);
            sc1 = wmma16(qf, kf1, sc1);
        }
        // ---- online softmax: per-row stats (row m = r + 8*hs, cols on lanes)
        float es[8];
#pragma unroll
        for (int r = 0; r < 8; ++r) {
            const float s0 = sc0[r] * alpha;
            const float s1 = sc1[r] * alpha;
            const float mx   = half_max(fmaxf(s0, s1));
            const float mnew = fmaxf(rmax[r], mx);
            es[r]   = __expf(rmax[r] - mnew);
            rmax[r] = mnew;
            const float p0 = __expf(s0 - mnew);
            const float p1 = __expf(s1 - mnew);
            rsum[r] = rsum[r] * es[r] + half_sum(p0 + p1);
            pw[(r + 8 * hs) * 32 + l15]      = (_Float16)p0;   // P[q][k] to LDS
            pw[(r + 8 * hs) * 32 + 16 + l15] = (_Float16)p1;
        }
#pragma unroll
        for (int t = 0; t < NT; ++t)
#pragma unroll
            for (int r = 0; r < 8; ++r) oacc[t][r] *= es[r];
        // ---- reload P in A-fragment layout (16x32) ----
        v16h pf = load_fragh(pw + l15 * 32 + hs * 16,
                             pw + l15 * 32 + hs * 16 + 8);
        // ---- O += P @ V (one WMMA per 16-col output tile) ----
#pragma unroll
        for (int t = 0; t < NT; ++t) {
            v16h vf = *(const v16h*)(Vb + (size_t)(t * 16 + l15) * S_LEN + k0 + hs * 16);
            oacc[t] = wmma16(pf, vf, oacc[t]);
        }
    }
    // ---- normalize + store ----
#pragma unroll
    for (int t = 0; t < NT; ++t) {
#pragma unroll
        for (int r = 0; r < 8; ++r) {
            const int q = q0 + r + 8 * hs;
            O[(size_t)q * DMODEL + h * HD + t * 16 + l15] =
                (_Float16)(oacc[t][r] / rsum[r]);
        }
    }
}

// ---------------- f32 -> f16 conversion (n multiple of 4) ----------------
__global__ void __launch_bounds__(256)
cvt_f16_k(const float* __restrict__ in, _Float16* __restrict__ out, int n4)
{
    const int i = blockIdx.x * 256 + threadIdx.x;
    if (i >= n4) return;
    float4 v = ((const float4*)in)[i];
    v4h hv;
    hv[0] = (_Float16)v.x; hv[1] = (_Float16)v.y;
    hv[2] = (_Float16)v.z; hv[3] = (_Float16)v.w;
    ((v4h*)out)[i] = hv;
}

// ---------------- LayerNorm: one wave per 512-wide row --------------------
__global__ void __launch_bounds__(256)
layernorm_k(const float* __restrict__ x, const float* __restrict__ g,
            const float* __restrict__ b,
            float* __restrict__ yf, _Float16* __restrict__ yh)
{
    const int lane = threadIdx.x & 31, wv = threadIdx.x >> 5;
    const int row = blockIdx.x * 8 + wv;
    const float* xr = x + (size_t)row * DMODEL;
    float v[16];
    float s = 0.0f;
#pragma unroll
    for (int e = 0; e < 16; ++e) { v[e] = xr[lane + 32 * e]; s += v[e]; }
    s = wave_sum(s);
    const float mean = s * (1.0f / DMODEL);
    float s2 = 0.0f;
#pragma unroll
    for (int e = 0; e < 16; ++e) { float d = v[e] - mean; s2 += d * d; }
    s2 = wave_sum(s2);
    const float inv = rsqrtf(s2 * (1.0f / DMODEL) + 1e-5f);
#pragma unroll
    for (int e = 0; e < 16; ++e) {
        const int c = lane + 32 * e;
        const float y = (v[e] - mean) * inv * g[c] + b[c];
        if (yf) yf[(size_t)row * DMODEL + c] = y;
        if (yh) yh[(size_t)row * DMODEL + c] = (_Float16)y;
    }
}

// ---------------- Fourier positional features (f16 out) -------------------
__global__ void __launch_bounds__(256)
fourier_k(const float* __restrict__ coords, _Float16* __restrict__ four)
{
    const int idx = blockIdx.x * 256 + threadIdx.x;     // S*128 threads
    const int s = idx >> 7;
    const int t = idx & 127;
    const int a = t >> 6;
    const int f = t & 63;
    const float c = coords[s * 2 + a];
    const float freq = exp2f(9.0f * (float)f / 63.0f);
    const float ang = (c * 2.0f - 1.0f) * freq * 3.14159265358979323846f;
    four[(size_t)s * 256 + a * 128 + f]      = (_Float16)sinf(ang);
    four[(size_t)s * 256 + a * 128 + 64 + f] = (_Float16)cosf(ang);
}

// ---------------- 32x32 LDS transpose (f16): dst[c][r] = src[r][soff+c] ---
__global__ void __launch_bounds__(256)
transpose_h_k(const _Float16* __restrict__ src, int lds_, int soff,
              _Float16* __restrict__ dst, int ldd)
{
    __shared__ _Float16 tile[32][34];
    const int tx = threadIdx.x, ty = threadIdx.y;       // (32,8)
    const int r0 = blockIdx.y * 32, c0 = blockIdx.x * 32;
#pragma unroll
    for (int i = 0; i < 4; ++i)
        tile[ty + i * 8][tx] = src[(size_t)(r0 + ty + i * 8) * lds_ + soff + c0 + tx];
    __syncthreads();
#pragma unroll
    for (int i = 0; i < 4; ++i)
        dst[(size_t)(c0 + ty + i * 8) * ldd + r0 + tx] = tile[tx][ty + i * 8];
}

// ---------------- Output head final dot: out[row*stride] = h·w + b --------
__global__ void __launch_bounds__(256)
head_dot_k(const float* __restrict__ h, const float* __restrict__ w,
           const float* __restrict__ bsc, float* __restrict__ out, int stride)
{
    const int lane = threadIdx.x & 31, wv = threadIdx.x >> 5;
    const int row = blockIdx.x * 8 + wv;
    const float* hr = h + (size_t)row * 256;
    float s = 0.0f;
#pragma unroll
    for (int e = 0; e < 8; ++e) { const int c = lane + 32 * e; s += hr[c] * w[c]; }
    s = wave_sum(s);
    if (lane == 0) out[(size_t)row * stride] = s + bsc[0];
}

// ---------------------------------------------------------------------------
// Host orchestration. Parameter flattening (dict insertion order):
//  0 features  1 coordinates
//  2,3 feat_proj.w/b  4,5 feat_ln.g/b  6,7 pos_l1.w/b  8,9 pos_l2.w/b
//  10.. light(2 layers x12) 34.. temp(2x12) 58.. walk(3x12)
//    layer: ln1.g,ln1.b,ln2.g,ln2.b,inp.w,inp.b,outp.w,outp.b,ff1.w,ff1.b,ff2.w,ff2.b
//  94.. heads: (l1.w,l1.b,l2.w,l2.b) x {light,temp_1,temp_2,walk_1..walk_5}
// d_out: light[4096] | temp[4096,2] | walk[4096,5]  (f32)
// ---------------------------------------------------------------------------
extern "C" void kernel_launch(void* const* d_in, const int* in_sizes, int n_in,
                              void* d_out, int out_size, void* d_ws, size_t ws_size,
                              hipStream_t stream)
{
    (void)in_sizes; (void)n_in; (void)out_size; (void)ws_size;
    const int S = S_LEN, D = DMODEL;
    auto P = [&](int i) { return (const float*)d_in[i]; };
    float* out = (float*)d_out;

    char* wsb = (char*)d_ws;
    size_t off = 0;
    auto falloc = [&](size_t elems) {
        float* p = (float*)(wsb + off);
        off += ((elems * 4) + 255) & ~(size_t)255;
        return p;
    };
    auto halloc = [&](size_t elems) {
        _Float16* p = (_Float16*)(wsb + off);
        off += ((elems * 2) + 255) & ~(size_t)255;
        return p;
    };

    // f32 buffers
    float* X  = falloc((size_t)S * D);        // stem activation / residual
    float* XL = falloc((size_t)S * D);        // per-branch residual stream
    float* HH = falloc((size_t)S * 256);      // head hidden
    // f16 buffers
    _Float16* FEATH = halloc((size_t)S * 256);    // features f16
    _Float16* FOUR  = halloc((size_t)S * 256);    // fourier features
    _Float16* PH    = halloc((size_t)S * D);      // pos mlp hidden
    _Float16* LNX   = halloc((size_t)S * D);      // layernorm output
    _Float16* T1    = halloc((size_t)S * 3 * D);  // qkv
    _Float16* T2    = halloc((size_t)S * 4 * D);  // ffn hidden
    _Float16* AO    = halloc((size_t)S * D);      // attention out
    _Float16* VT    = halloc((size_t)S * D);      // V transposed [D,S]
    _Float16* XH    = halloc((size_t)S * D);      // branch output f16 (for heads)

    auto cvt = [&](const float* src, _Float16* dst, size_t elems) {
        const int n4 = (int)(elems / 4);
        cvt_f16_k<<<dim3((unsigned)((n4 + 255) / 256)), dim3(256), 0, stream>>>(src, dst, n4);
    };
    auto cvtW = [&](int idx, size_t elems) {
        _Float16* p = halloc(elems);
        cvt(P(idx), p, elems);
        return p;
    };

    // ---- one-shot f16 conversions (weights + feature input) ----
    cvt(P(0), FEATH, (size_t)S * 256);
    _Float16* Wfeat = cvtW(2, 512 * 256);
    _Float16* Wp1   = cvtW(6, 512 * 256);
    _Float16* Wp2   = cvtW(8, 512 * 512);
    const int bbase[3]  = {10, 34, 58};
    const int bnlay[3]  = {2, 2, 3};
    const int bnhead[3] = {2, 2, 4};
    const int bheads[3] = {1, 2, 5};
    _Float16* Wlay[7][4];   // per global layer: in_proj, out_proj, ff1, ff2
    {
        int gl = 0;
        for (int bi = 0; bi < 3; ++bi)
            for (int l = 0; l < bnlay[bi]; ++l, ++gl) {
                const int pb = bbase[bi] + l * 12;
                Wlay[gl][0] = cvtW(pb + 4,  (size_t)1536 * 512);
                Wlay[gl][1] = cvtW(pb + 6,  (size_t)512 * 512);
                Wlay[gl][2] = cvtW(pb + 8,  (size_t)2048 * 512);
                Wlay[gl][3] = cvtW(pb + 10, (size_t)512 * 2048);
            }
    }
    _Float16* Whead[8];
    for (int i = 0; i < 8; ++i) Whead[i] = cvtW(94 + i * 4, (size_t)256 * 512);

    auto gemm = [&](const _Float16* A, int lda, const _Float16* B, int ldb,
                    const float* bias, const float* resid, int ldr,
                    float* Cf, _Float16* Ch, int ldc,
                    int M, int N, int K, float alpha, int act) {
        dim3 g((unsigned)(N / 64), (unsigned)(M / 256));
        gemm_nt_wmma_h<<<g, dim3(256), 0, stream>>>(A, lda, B, ldb, bias,
                                                    resid, ldr, Cf, Ch, ldc,
                                                    M, N, K, alpha, act);
    };

    // ---- stem: x = LN(features @ Wf^T + bf) + posMLP(fourier(coords)) ----
    gemm(FEATH, 256, Wfeat, 256, P(3), nullptr, 0, X, nullptr, D, S, D, 256, 1.0f, 0);
    layernorm_k<<<dim3(S / 8), dim3(256), 0, stream>>>(X, P(4), P(5), X, nullptr);
    fourier_k<<<dim3(S * 128 / 256), dim3(256), 0, stream>>>(P(1), FOUR);
    gemm(FOUR, 256, Wp1, 256, P(7), nullptr, 0, nullptr, PH, D, S, D, 256, 1.0f, 2);
    gemm(PH, D, Wp2, D, P(9), X, D, X, nullptr, D, S, D, D, 1.0f, 0);   // x += pos

    float* obase[3]      = {out, out + S, out + 3 * S};
    const int ostride[3] = {1, 2, 5};
    int headIdx = 0, glayer = 0;

    for (int bi = 0; bi < 3; ++bi) {
        hipMemcpyAsync(XL, X, (size_t)S * D * 4, hipMemcpyDeviceToDevice, stream);
        const int nh = bnhead[bi], hd = D / nh;
        const float isq = 1.0f / sqrtf((float)hd);

        for (int l = 0; l < bnlay[bi]; ++l, ++glayer) {
            const int pb = bbase[bi] + l * 12;
            // --- pre-LN MHA ---
            layernorm_k<<<dim3(S / 8), dim3(256), 0, stream>>>(XL, P(pb + 0), P(pb + 1),
                                                               nullptr, LNX);
            gemm(LNX, D, Wlay[glayer][0], D, P(pb + 5), nullptr, 0,
                 nullptr, T1, 3 * D, S, 3 * D, D, 1.0f, 0);            // qkv (f16)
            transpose_h_k<<<dim3(D / 32, S / 32), dim3(32, 8), 0, stream>>>(
                T1, 3 * D, 2 * D, VT, S);                              // VT[d][s]=V[s][d]
            if (nh == 2)
                flash_attn_k<256><<<dim3(S / 128, 2), dim3(256), 0, stream>>>(
                    T1, VT, AO, isq);
            else
                flash_attn_k<128><<<dim3(S / 128, 4), dim3(256), 0, stream>>>(
                    T1, VT, AO, isq);
            gemm(AO, D, Wlay[glayer][1], D, P(pb + 7), XL, D,
                 XL, nullptr, D, S, D, D, 1.0f, 0);                    // out_proj + resid
            // --- pre-LN FFN ---
            layernorm_k<<<dim3(S / 8), dim3(256), 0, stream>>>(XL, P(pb + 2), P(pb + 3),
                                                               nullptr, LNX);
            gemm(LNX, D, Wlay[glayer][2], D, P(pb + 9), nullptr, 0,
                 nullptr, T2, 4 * D, S, 4 * D, D, 1.0f, 1);            // ff1 + relu (f16)
            gemm(T2, 4 * D, Wlay[glayer][3], 4 * D, P(pb + 11), XL, D,
                 XL, nullptr, D, S, D, 4 * D, 1.0f, 0);                // ff2 + resid
        }

        // --- output heads for this branch ---
        cvt(XL, XH, (size_t)S * D);
        for (int k = 0; k < bheads[bi]; ++k) {
            const int hb = 94 + headIdx * 4; ++headIdx;
            gemm(XH, D, Whead[headIdx - 1], D, P(hb + 1), nullptr, 0,
                 HH, nullptr, 256, S, 256, D, 1.0f, 2);                // l1 + gelu
            head_dot_k<<<dim3(S / 8), dim3(256), 0, stream>>>(
                HH, P(hb + 2), P(hb + 3), obase[bi] + k, ostride[bi]);
        }
    }
}
